// Attention_87290915324637
// MI455X (gfx1250) — matrix operand order repaired and checked by reading, not compiled, not run on hardware
//
#include <hip/hip_runtime.h>
#include <hip/hip_bf16.h>

// ---------------------------------------------------------------------------
// Causal attention for B=8, T=2048, C=768, D=64 on gfx1250 (MI455X).
// Compute-bound -> everything through v_wmma_f32_16x16x32_bf16 (f32 accum).
//
// Kernel 0: one-shot f32 -> bf16 conversion of x and Wq/Wk/Wv into d_ws
//           (keeps conversions OUT of the GEMM hot loop).
// Kernel 1: QKV projection GEMM [16384x768]x[768x64], pure-bf16 WMMA loop,
//           writes Q[b][t][64], Kt[b][64][t] (transposed), V[b][t][64].
// Kernel 2: flash-attention, 1 wave per 16-row query tile, key blocks of 32,
//           online softmax in C-layout registers, P transposed via LDS into
//           the 16-bit WMMA A-fragment layout.
// ---------------------------------------------------------------------------

typedef __attribute__((ext_vector_type(16))) __bf16 v16bf;
typedef __attribute__((ext_vector_type(8)))  __bf16 v8bf;
typedef __attribute__((ext_vector_type(8)))  float  v8f;

#define WMMA_BF16(a, b, c) \
  __builtin_amdgcn_wmma_f32_16x16x32_bf16(false, (a), false, (b), (short)0, (c), false, false)

#define BATCH 8
#define SEQ   2048
#define CIN   768
#define DOUT  64

__device__ __forceinline__ v8f zero8() {
  v8f z;
#pragma unroll
  for (int i = 0; i < 8; ++i) z[i] = 0.0f;
  return z;
}

__device__ __forceinline__ v16bf pack16(v8bf lo, v8bf hi) {
  v16bf r;
#pragma unroll
  for (int i = 0; i < 8; ++i) { r[i] = lo[i]; r[i + 8] = hi[i]; }
  return r;
}

// Butterfly reductions within each 16-lane half of the wave32
// (masks 1/2/4/8 never cross the 16-lane boundary).
__device__ __forceinline__ float half_reduce_max(float v) {
#pragma unroll
  for (int m = 8; m >= 1; m >>= 1) v = fmaxf(v, __shfl_xor(v, m, 32));
  return v;
}
__device__ __forceinline__ float half_reduce_sum(float v) {
#pragma unroll
  for (int m = 8; m >= 1; m >>= 1) v += __shfl_xor(v, m, 32);
  return v;
}

// ---------------------------------------------------------------------------
// Kernel 0: bulk f32 -> bf16 conversion, 8 elements per thread.
// ---------------------------------------------------------------------------
__global__ __launch_bounds__(256) void cvt_bf16_kernel(
    const float* __restrict__ in, __bf16* __restrict__ outp, int n8)
{
  const int i = blockIdx.x * 256 + threadIdx.x;
  if (i < n8) {
    v8f f = *(const v8f*)(in + (size_t)i * 8);
    *(v8bf*)(outp + (size_t)i * 8) = __builtin_convertvector(f, v8bf);
  }
}

// ---------------------------------------------------------------------------
// Kernel 1: QKV projections. One wave computes one 16-row x 64-col tile of
// one of {Q, K, V}. 3072 waves total = 768 blocks x 4 waves.
//
// 16-bit A-fragment (16x32): lane l holds row M = l%16; half h holds
//   K = (h<8 ? 0 : 16) + (l>=16 ? 8 : 0) + h%8      -> two 16-B loads.
// B-fragment (32x16): lane%16 = N column, halves = K in the A-fragment order
//                     -> 16 strided 2-B loads (stride DOUT).
// C/D (16x16 f32): VGPR r, lane l -> M = r + (l>=16 ? 8 : 0), N = l%16.
// ---------------------------------------------------------------------------
__global__ __launch_bounds__(128) void qkv_proj_kernel(
    const __bf16* __restrict__ xb, const __bf16* __restrict__ Wb,
    __bf16* __restrict__ Qo, __bf16* __restrict__ Kt, __bf16* __restrict__ Vo)
{
  const int wave  = blockIdx.x * 4 + (threadIdx.x >> 5);
  const int lane  = threadIdx.x & 31;
  const int laneN = lane & 15;
  const int hi    = lane >> 4;

  const int mat = wave >> 10;        // 0=Q, 1=K, 2=V (1024 row-tiles each)
  const int rt  = wave & 1023;
  const int b   = rt >> 7;           // 128 row-tiles per batch
  const int t0  = (rt & 127) << 4;

  v8f acc[4];
#pragma unroll
  for (int j = 0; j < 4; ++j) acc[j] = zero8();

  const __bf16* xrow = xb + (size_t)((b << 11) + t0 + laneN) * CIN;
  const __bf16* Wm   = Wb + (size_t)mat * CIN * DOUT;

  for (int kc = 0; kc < CIN / 32; ++kc) {
    const int abase = (kc << 5) + (hi << 3);
    v16bf a = pack16(*(const v8bf*)(xrow + abase),
                     *(const v8bf*)(xrow + abase + 16));
    const __bf16* wrow = Wm + (size_t)(kc << 5) * DOUT + laneN;
#pragma unroll
    for (int j = 0; j < 4; ++j) {
      v16bf bw;
#pragma unroll
      for (int s2 = 0; s2 < 16; ++s2)
        bw[s2] = wrow[(size_t)((s2 & 7) + (hi << 3) + ((s2 >> 3) << 4)) * DOUT + 16 * j];
      acc[j] = WMMA_BF16(a, bw, acc[j]);
    }
  }

  if (mat == 1) {
    // K stored transposed: Kt[b][d][t] so QK^T B-fragments load contiguously.
#pragma unroll
    for (int j = 0; j < 4; ++j)
#pragma unroll
      for (int r = 0; r < 8; ++r) {
        const int d = 16 * j + laneN;
        const int t = t0 + r + (hi << 3);
        Kt[((size_t)(b * DOUT + d) << 11) + t] = (__bf16)acc[j][r];
      }
  } else {
    __bf16* O = (mat == 0) ? Qo : Vo;
#pragma unroll
    for (int j = 0; j < 4; ++j)
#pragma unroll
      for (int r = 0; r < 8; ++r)
        O[(size_t)((b << 11) + t0 + r + (hi << 3)) * DOUT + 16 * j + laneN]
            = (__bf16)acc[j][r];
  }
}

// ---------------------------------------------------------------------------
// Kernel 2: flash attention. One wave owns 16 query rows and the full D=64
// output (4x v8f accumulators). Key blocks of 32: S = Q*K^T (4 WMMAs),
// online softmax, P through per-wave LDS tile into A-fragment layout,
// O += P*V (4 WMMAs).
// ---------------------------------------------------------------------------
__global__ __launch_bounds__(128) void flash_attn_kernel(
    const __bf16* __restrict__ Q, const __bf16* __restrict__ Kt,
    const __bf16* __restrict__ V, float* __restrict__ out)
{
  // 16 rows x 32 cols bf16 per wave, row stride 48 halves (96 B, 16B-aligned).
  __shared__ __attribute__((aligned(32))) __bf16 sP[4][16 * 48];

  const int wv    = threadIdx.x >> 5;
  const int lane  = threadIdx.x & 31;
  const int laneN = lane & 15;
  const int hi    = lane >> 4;
  const int wave  = blockIdx.x * 4 + wv;
  const int b     = wave >> 7;
  const int qr0   = (wave & 127) << 4;

  // Q A-fragments for both 32-wide d-chunks, loaded once.
  const __bf16* qrow = Q + (size_t)((b << 11) + qr0 + laneN) * DOUT;
  v16bf qa[2];
#pragma unroll
  for (int c = 0; c < 2; ++c) {
    const int base = (c << 5) + (hi << 3);
    qa[c] = pack16(*(const v8bf*)(qrow + base), *(const v8bf*)(qrow + base + 16));
  }

  v8f acc[4];
#pragma unroll
  for (int j = 0; j < 4; ++j) acc[j] = zero8();

  float m_i[8], l_i[8];
#pragma unroll
  for (int r = 0; r < 8; ++r) { m_i[r] = -1e30f; l_i[r] = 0.0f; }

  const int nblocks = (qr0 >> 5) + 1;  // causal: keys 0 .. qr0+15
  for (int blk = 0; blk < nblocks; ++blk) {
    const int kb = blk << 5;

    // ---- S = Q * K^T : two 16-key column tiles, K-dim 64 = 2 chunks ----
    v8f s[2];
#pragma unroll
    for (int j = 0; j < 2; ++j) {
      v8f z = zero8();
#pragma unroll
      for (int c = 0; c < 2; ++c) {
        // lane%16 = key column; elements = d in A-fragment k order.
        v16bf bk;
#pragma unroll
        for (int s2 = 0; s2 < 16; ++s2)
          bk[s2] = Kt[((size_t)(b * DOUT + (c << 5) + (s2 & 7) + (hi << 3) + ((s2 >> 3) << 4)) << 11) + kb + 16 * j + laneN];
        z = WMMA_BF16(qa[c], bk, z);
      }
      s[j] = z;
    }

    // ---- scale, causal mask, online softmax (per row r of this half) ----
#pragma unroll
    for (int r = 0; r < 8; ++r) {
      const int qg = qr0 + r + (hi << 3);
      float s0 = s[0][r] * 0.125f;           // D^-0.5 = 1/8
      float s1 = s[1][r] * 0.125f;
      if (kb + laneN > qg)      s0 = -1e30f;
      if (kb + 16 + laneN > qg) s1 = -1e30f;

      float bm    = half_reduce_max(fmaxf(s0, s1));
      float mn    = fmaxf(m_i[r], bm);
      float alpha = __expf(m_i[r] - mn);
      float p0    = __expf(s0 - mn);
      float p1    = __expf(s1 - mn);
      float rs    = half_reduce_sum(p0 + p1);
      l_i[r] = l_i[r] * alpha + rs;
      m_i[r] = mn;
      s[0][r] = p0;
      s[1][r] = p1;
#pragma unroll
      for (int j2 = 0; j2 < 4; ++j2) acc[j2][r] *= alpha;
    }

    // ---- transpose P (C-layout regs) -> A-fragment layout via LDS ----
#pragma unroll
    for (int j = 0; j < 2; ++j)
#pragma unroll
      for (int r = 0; r < 8; ++r)
        sP[wv][(r + (hi << 3)) * 48 + 16 * j + laneN] = (__bf16)s[j][r];

    asm volatile("s_wait_dscnt 0x0" ::: "memory");  // same-wave LDS RAW

    v16bf pa = pack16(*(const v8bf*)&sP[wv][laneN * 48 + (hi << 3)],
                      *(const v8bf*)&sP[wv][laneN * 48 + (hi << 3) + 16]);

    // ---- O += P * V : lane%16 = d column, elements = keys in A-fragment k order ----
    const __bf16* vrow = V + (size_t)((b << 11) + kb) * DOUT + laneN;
#pragma unroll
    for (int j2 = 0; j2 < 4; ++j2) {
      v16bf bv;
#pragma unroll
      for (int s2 = 0; s2 < 16; ++s2)
        bv[s2] = vrow[(size_t)((s2 & 7) + (hi << 3) + ((s2 >> 3) << 4)) * DOUT + 16 * j2];
      acc[j2] = WMMA_BF16(pa, bv, acc[j2]);
    }
  }

  // ---- normalize and store fp32 output [b][t][64] ----
#pragma unroll
  for (int r = 0; r < 8; ++r) {
    const float rinv = 1.0f / l_i[r];
#pragma unroll
    for (int j = 0; j < 4; ++j)
      out[(size_t)((b << 11) + qr0 + r + (hi << 3)) * DOUT + 16 * j + laneN]
          = acc[j][r] * rinv;
  }
}

// ---------------------------------------------------------------------------
extern "C" void kernel_launch(void* const* d_in, const int* in_sizes, int n_in,
                              void* d_out, int out_size, void* d_ws, size_t ws_size,
                              hipStream_t stream) {
  const float* x  = (const float*)d_in[0];
  const float* Wq = (const float*)d_in[1];
  const float* Wk = (const float*)d_in[2];
  const float* Wv = (const float*)d_in[3];

  // Workspace layout (bf16 elements):
  //   Qb   [8*2048*64]  = 1,048,576
  //   Ktb  [8*64*2048]  = 1,048,576
  //   Vb   [8*2048*64]  = 1,048,576
  //   xb   [8*2048*768] = 12,582,912
  //   Wb   [3*768*64]   = 147,456        (~31.8 MB total)
  const size_t QKV_ELTS = (size_t)BATCH * SEQ * DOUT;
  const size_t X_ELTS   = (size_t)BATCH * SEQ * CIN;
  const size_t W_ELTS   = (size_t)CIN * DOUT;

  __bf16* Qb  = (__bf16*)d_ws;
  __bf16* Ktb = Qb  + QKV_ELTS;
  __bf16* Vb  = Ktb + QKV_ELTS;
  __bf16* xbb = Vb  + QKV_ELTS;
  __bf16* Wbb = xbb + X_ELTS;

  // One-shot f32 -> bf16 conversions (keeps converts out of the GEMM loop).
  {
    const int n8 = (int)(X_ELTS / 8);                 // 1,572,864
    cvt_bf16_kernel<<<(n8 + 255) / 256, 256, 0, stream>>>(x, xbb, n8);
    const int w8 = (int)(W_ELTS / 8);                 // 6,144
    cvt_bf16_kernel<<<(w8 + 255) / 256, 256, 0, stream>>>(Wq, Wbb,             w8);
    cvt_bf16_kernel<<<(w8 + 255) / 256, 256, 0, stream>>>(Wk, Wbb + W_ELTS,    w8);
    cvt_bf16_kernel<<<(w8 + 255) / 256, 256, 0, stream>>>(Wv, Wbb + 2*W_ELTS,  w8);
  }

  // 3 matrices x 1024 row-tiles = 3072 waves; 4 waves per 128-thread block.
  qkv_proj_kernel<<<768, 128, 0, stream>>>(xbb, Wbb, Qb, Ktb, Vb);

  // 1024 query tiles (8 batches x 128 tiles); 4 waves per block.
  flash_attn_kernel<<<256, 128, 0, stream>>>(Qb, Ktb, Vb, (float*)d_out);
}
